// GPT_72653666779355
// MI455X (gfx1250) — matrix operand order repaired and checked by reading, not compiled, not run on hardware
//
#include <hip/hip_runtime.h>

typedef _Float16 h16;
typedef __attribute__((ext_vector_type(8)))  _Float16 half8;
typedef __attribute__((ext_vector_type(16))) _Float16 v16h;
typedef __attribute__((ext_vector_type(8)))  float    v8f;

constexpr int Bc  = 2;
constexpr int Tc  = 2048;
constexpr int Cc  = 1024;
constexpr int NHc = 16;
constexpr int NKVc = 8;
constexpr int HDc = 64;
constexpr int Mc  = Bc * Tc;   // 4096 rows
constexpr int WINc = 1024;

__device__ __forceinline__ v16h pack16(half8 lo, half8 hi) {
  union { v16h v; half8 h[2]; } u;
  u.h[0] = lo; u.h[1] = hi;
  return u.v;
}
__device__ __forceinline__ half8 ld8(const h16* p) { return *(const half8*)p; }

// ---------------------------------------------------------------------------
// fp32 -> f16 convert
// ---------------------------------------------------------------------------
__global__ __launch_bounds__(256) void cvt_f16(const float* __restrict__ in,
                                               h16* __restrict__ out, int n) {
  int i = blockIdx.x * 256 + threadIdx.x;
  if (i < n) out[i] = (h16)in[i];
}

// ---------------------------------------------------------------------------
// fp32 (K x N) -> f16 transposed (N x K): makes WMMA B-operand K-runs contiguous
// ---------------------------------------------------------------------------
__global__ __launch_bounds__(256) void transpose_cvt(const float* __restrict__ in,
                                                     h16* __restrict__ out,
                                                     int K, int N) {
  int i = blockIdx.x * 256 + threadIdx.x;
  if (i >= K * N) return;
  int k = i / N, n = i % N;
  out[(size_t)n * K + k] = (h16)in[i];
}

// ---------------------------------------------------------------------------
// WMMA GEMM: C(f32, MxN) = A(f16, MxK) * Bt(f16, NxK)^T
// 8 waves per block; each wave computes a 32x64 tile (2 m-tiles x 4 n-tiles)
// so each B fragment feeds two v_wmma_f32_16x16x32_f16 -> ~21 flop/byte vs L2.
// ---------------------------------------------------------------------------
__global__ __launch_bounds__(256) void gemm16(const h16* __restrict__ A,
                                              const h16* __restrict__ Bt,
                                              float* __restrict__ Cmat,
                                              int M, int N, int K) {
  const int wave  = threadIdx.x >> 5;
  const int lane  = threadIdx.x & 31;
  const int half_ = lane >> 4;
  const int l15   = lane & 15;
  const int mt0   = (blockIdx.x * 8 + wave) * 2;   // two 16-row tiles per wave
  const int nbase = blockIdx.y * 64;
  if (mt0 * 16 >= M) return;

  const h16* arow0 = A + (size_t)(mt0 * 16 + l15) * K;
  const h16* arow1 = arow0 + (size_t)16 * K;
  v8f acc0[4] = {v8f{}, v8f{}, v8f{}, v8f{}};
  v8f acc1[4] = {v8f{}, v8f{}, v8f{}, v8f{}};

  for (int kb = 0; kb < K; kb += 32) {
    // pull the A streams ahead (global_prefetch_b8)
    __builtin_prefetch(arow0 + kb + 256, 0, 1);
    __builtin_prefetch(arow1 + kb + 256, 0, 1);
    // A 16x32 f16 layout: lane half 0 -> K {0..7, 16..23}, half 1 -> {8..15, 24..31}
    v16h a0 = pack16(ld8(arow0 + kb + half_ * 8), ld8(arow0 + kb + 16 + half_ * 8));
    v16h a1 = pack16(ld8(arow1 + kb + half_ * 8), ld8(arow1 + kb + 16 + half_ * 8));
    const int krb = kb + half_ * 8;  // B layout: half 0 -> K {0..7, 16..23}, half 1 -> {8..15, 24..31}
#pragma unroll
    for (int s = 0; s < 4; ++s) {
      const h16* bcol = Bt + (size_t)(nbase + s * 16 + l15) * K + krb;
      v16h b = pack16(ld8(bcol), ld8(bcol + 16));
      acc0[s] = __builtin_amdgcn_wmma_f32_16x16x32_f16(false, a0, false, b,
                                                       (short)0, acc0[s], false, false);
      acc1[s] = __builtin_amdgcn_wmma_f32_16x16x32_f16(false, a1, false, b,
                                                       (short)0, acc1[s], false, false);
    }
  }
  // C/D layout: m = j + 8*half, n = lane&15
  const int m0 = mt0 * 16 + 8 * half_;
#pragma unroll
  for (int s = 0; s < 4; ++s)
#pragma unroll
    for (int j = 0; j < 8; ++j) {
      Cmat[(size_t)(m0 + j) * N + nbase + s * 16 + l15]      = acc0[s][j];
      Cmat[(size_t)(m0 + 16 + j) * N + nbase + s * 16 + l15] = acc1[s][j];
    }
}

// ---------------------------------------------------------------------------
// Pointwise: gate/value-embed add, RoPE, RMS-norm; emits f16 tensors laid out
// for the attention kernel's WMMA operands.
//   qh: (b, h,  t, d)   kh: (b, kvh, t, d)   vt: (b, kvh, d, t)  [transposed!]
// ---------------------------------------------------------------------------
__global__ __launch_bounds__(128) void pointwise(const float* __restrict__ x,
                                                 const float* __restrict__ ve,
                                                 const float* __restrict__ cosb,
                                                 const float* __restrict__ sinb,
                                                 const float* __restrict__ Wg,
                                                 const float* __restrict__ qf,
                                                 const float* __restrict__ kf,
                                                 const float* __restrict__ vf,
                                                 h16* __restrict__ qh,
                                                 h16* __restrict__ kh,
                                                 h16* __restrict__ vt) {
  const int row = blockIdx.x;        // b*T + t
  const int b = row / Tc, t = row % Tc;
  const int tid = threadIdx.x;

  __shared__ float gate[NKVc];
  if (tid < NKVc) {
    float s = 0.f;
    for (int c = 0; c < 32; ++c) s += x[(size_t)row * Cc + c] * Wg[c * NKVc + tid];
    gate[tid] = 2.f / (1.f + expf(-s));   // 2*sigmoid
  }
  __syncthreads();

  // v = v + gate * ve, stored transposed (d-major over t) for P@V B-operand
#pragma unroll
  for (int i = 0; i < 4; ++i) {
    int idx = tid + 128 * i;
    int hd = idx >> 6, d = idx & 63;
    float v = vf[(size_t)row * 512 + idx] + gate[hd] * ve[(size_t)row * 512 + idx];
    vt[((size_t)(b * NKVc + hd) * HDc + d) * Tc + t] = (h16)v;
  }

  const int d = tid & 31;                       // rotary dim 0..31 (pair with d+32)
  const float cs = cosb[t * 32 + d];
  const float sn = sinb[t * 32 + d];

  // Q: rope + rmsnorm (head sumsq = wave-wide reduction, all lanes share a head)
#pragma unroll
  for (int i = 0; i < 4; ++i) {
    int hd = (tid >> 5) + 4 * i;
    float x1 = qf[(size_t)row * 1024 + hd * 64 + d];
    float x2 = qf[(size_t)row * 1024 + hd * 64 + d + 32];
    float o1 = x1 * cs + x2 * sn;
    float o2 = -x1 * sn + x2 * cs;
    float ss = o1 * o1 + o2 * o2;
    for (int m = 1; m < 32; m <<= 1) ss += __shfl_xor(ss, m, 32);
    float rn = rsqrtf(ss * (1.f / 64.f) + 1.1920929e-7f);
    size_t base = ((size_t)(b * NHc + hd) * Tc + t) * HDc;
    qh[base + d]      = (h16)(o1 * rn);
    qh[base + d + 32] = (h16)(o2 * rn);
  }
  // K: same
#pragma unroll
  for (int i = 0; i < 2; ++i) {
    int hd = (tid >> 5) + 4 * i;
    float x1 = kf[(size_t)row * 512 + hd * 64 + d];
    float x2 = kf[(size_t)row * 512 + hd * 64 + d + 32];
    float o1 = x1 * cs + x2 * sn;
    float o2 = -x1 * sn + x2 * cs;
    float ss = o1 * o1 + o2 * o2;
    for (int m = 1; m < 32; m <<= 1) ss += __shfl_xor(ss, m, 32);
    float rn = rsqrtf(ss * (1.f / 64.f) + 1.1920929e-7f);
    size_t base = ((size_t)(b * NKVc + hd) * Tc + t) * HDc;
    kh[base + d]      = (h16)(o1 * rn);
    kh[base + d + 32] = (h16)(o2 * rn);
  }
}

// ---------------------------------------------------------------------------
// Flash attention: one wave per 16-query tile. Heads 0..7 causal w/ kv 0..3,
// heads 8..15 sliding-window-1024 w/ kv 4..7 (GQA group size 2).
// yh out: (b, t, h*64+d) f16 == input rows of the projection GEMM.
// ---------------------------------------------------------------------------
__global__ __launch_bounds__(256) void attn(const h16* __restrict__ qh,
                                            const h16* __restrict__ kh,
                                            const h16* __restrict__ vt,
                                            h16* __restrict__ yh) {
  __shared__ h16 pbuf[8][16 * 32];
  const int wave  = threadIdx.x >> 5;
  const int lane  = threadIdx.x & 31;
  const int half_ = lane >> 4;
  const int l15   = lane & 15;
  const int qt = blockIdx.x * 8 + wave;     // 0..127
  const int bh = blockIdx.y;                // b*16 + h
  const int b = bh >> 4, h = bh & 15;
  const bool local = (h >= 8);
  const int kvh = local ? 4 + ((h - 8) >> 1) : (h >> 1);

  const h16* Q = qh + (size_t)(b * NHc + h) * Tc * HDc;
  const h16* K = kh + (size_t)(b * NKVc + kvh) * Tc * HDc;
  const h16* V = vt + (size_t)(b * NKVc + kvh) * HDc * Tc;

  // Q A-fragments (16 rows x 64 head-dim -> two K=32 fragments)
  const h16* qrow = Q + (size_t)(qt * 16 + l15) * HDc;
  const v16h aq0 = pack16(ld8(qrow + half_ * 8),      ld8(qrow + 16 + half_ * 8));
  const v16h aq1 = pack16(ld8(qrow + 32 + half_ * 8), ld8(qrow + 48 + half_ * 8));

  float rmax[8], rsum[8];
#pragma unroll
  for (int j = 0; j < 8; ++j) { rmax[j] = -1e30f; rsum[j] = 0.f; }
  v8f o[4] = {v8f{}, v8f{}, v8f{}, v8f{}};

  const int qlo = qt * 16;
  int lo = local ? (qlo - WINc) : 0;
  if (lo < 0) lo = 0;
  lo &= ~31;
  h16* pp = pbuf[wave];

  for (int kb = lo; kb < qlo + 16; kb += 32) {
    // ---- scores: two 16-key subtiles, each = Q(16x64) . K^T(64x16)
    v8f s0{}, s1{};
    {
      const h16* krow = K + (size_t)(kb + l15) * HDc;
      v16h bk0 = pack16(ld8(krow + half_ * 8),      ld8(krow + 16 + half_ * 8));
      v16h bk1 = pack16(ld8(krow + 32 + half_ * 8), ld8(krow + 48 + half_ * 8));
      s0 = __builtin_amdgcn_wmma_f32_16x16x32_f16(false, aq0, false, bk0, (short)0, s0, false, false);
      s0 = __builtin_amdgcn_wmma_f32_16x16x32_f16(false, aq1, false, bk1, (short)0, s0, false, false);
    }
    {
      const h16* krow = K + (size_t)(kb + 16 + l15) * HDc;
      v16h bk0 = pack16(ld8(krow + half_ * 8),      ld8(krow + 16 + half_ * 8));
      v16h bk1 = pack16(ld8(krow + 32 + half_ * 8), ld8(krow + 48 + half_ * 8));
      s1 = __builtin_amdgcn_wmma_f32_16x16x32_f16(false, aq0, false, bk0, (short)0, s1, false, false);
      s1 = __builtin_amdgcn_wmma_f32_16x16x32_f16(false, aq1, false, bk1, (short)0, s1, false, false);
    }

    // ---- online softmax over the 32-key block (C-layout: m=j+8*half, n=l15)
    const int k0 = kb + l15, k1 = kb + 16 + l15;
#pragma unroll
    for (int j = 0; j < 8; ++j) {
      const int q = qlo + 8 * half_ + j;
      float v0 = s0[j] * 0.125f;
      float v1 = s1[j] * 0.125f;
      if (!((k0 <= q) && (!local || (q - k0) <= WINc))) v0 = -1e30f;
      if (!((k1 <= q) && (!local || (q - k1) <= WINc))) v1 = -1e30f;
      float mx = fmaxf(v0, v1);
#pragma unroll
      for (int m = 1; m < 16; m <<= 1) mx = fmaxf(mx, __shfl_xor(mx, m, 16));
      const float newm = fmaxf(rmax[j], mx);
      const float corr = expf(rmax[j] - newm);
      float p0 = (v0 < -5e29f) ? 0.f : expf(v0 - newm);
      float p1 = (v1 < -5e29f) ? 0.f : expf(v1 - newm);
      float ps = p0 + p1;
#pragma unroll
      for (int m = 1; m < 16; m <<= 1) ps += __shfl_xor(ps, m, 16);
      rsum[j] = rsum[j] * corr + ps;
      rmax[j] = newm;
#pragma unroll
      for (int s = 0; s < 4; ++s) o[s][j] *= corr;
      // stage P into LDS in plain [m][k] order for A-operand regather
      pp[(8 * half_ + j) * 32 + l15]      = (h16)p0;
      pp[(8 * half_ + j) * 32 + 16 + l15] = (h16)p1;
    }
    asm volatile("s_wait_dscnt 0x0" ::: "memory");  // cross-lane LDS W->R ordering

    // P as A-operand: row l15, K runs {8*half.., 16+8*half..}
    const v16h ap = pack16(*(const half8*)(pp + l15 * 32 + half_ * 8),
                           *(const half8*)(pp + l15 * 32 + 16 + half_ * 8));
    // ---- O += P(16x32) . V(32x64): V transposed so key-runs are contiguous
#pragma unroll
    for (int s = 0; s < 4; ++s) {
      const h16* vrow = V + (size_t)(s * 16 + l15) * Tc + kb + half_ * 8;
      v16h bv = pack16(ld8(vrow), ld8(vrow + 16));
      o[s] = __builtin_amdgcn_wmma_f32_16x16x32_f16(false, ap, false, bv, (short)0, o[s], false, false);
    }
  }

  // epilogue: normalize and store into projection-GEMM input layout
#pragma unroll
  for (int s = 0; s < 4; ++s)
#pragma unroll
    for (int j = 0; j < 8; ++j) {
      const int q = qlo + 8 * half_ + j;
      yh[((size_t)(b * Tc + q)) * Cc + h * HDc + s * 16 + l15] = (h16)(o[s][j] / rsum[j]);
    }
}

// ---------------------------------------------------------------------------
extern "C" void kernel_launch(void* const* d_in, const int* in_sizes, int n_in,
                              void* d_out, int out_size, void* d_ws, size_t ws_size,
                              hipStream_t stream) {
  (void)in_sizes; (void)n_in; (void)out_size; (void)ws_size;
  const float* x    = (const float*)d_in[0];
  const float* ve   = (const float*)d_in[1];
  const float* cosb = (const float*)d_in[2];
  const float* sinb = (const float*)d_in[3];
  const float* Wq   = (const float*)d_in[4];
  const float* Wk   = (const float*)d_in[5];
  const float* Wv   = (const float*)d_in[6];
  const float* Wpr  = (const float*)d_in[7];
  const float* Wg   = (const float*)d_in[8];
  // d_in[9] = window scalar; compile-time constant (1024) for this shape.

  char* ws = (char*)d_ws;
  size_t off = 0;
  auto take = [&](size_t bytes) -> char* {
    char* p = ws + off;
    off = (off + bytes + 255) & ~(size_t)255;
    return p;
  };
  h16*   xh  = (h16*)take((size_t)Mc * Cc * 2);
  h16*   wqt = (h16*)take((size_t)Cc * Cc * 2);
  h16*   wkt = (h16*)take((size_t)512 * Cc * 2);
  h16*   wvt = (h16*)take((size_t)512 * Cc * 2);
  h16*   wpt = (h16*)take((size_t)Cc * Cc * 2);
  float* qf  = (float*)take((size_t)Mc * 1024 * 4);
  float* kf  = (float*)take((size_t)Mc * 512 * 4);
  float* vf  = (float*)take((size_t)Mc * 512 * 4);
  h16*   qhp = (h16*)take((size_t)Mc * 1024 * 2);
  h16*   khp = (h16*)take((size_t)Mc * 512 * 2);
  h16*   vtp = (h16*)take((size_t)Mc * 512 * 2);
  h16*   yh  = (h16*)take((size_t)Mc * Cc * 2);

  // precision prep
  cvt_f16<<<(Mc * Cc + 255) / 256, 256, 0, stream>>>(x, xh, Mc * Cc);
  transpose_cvt<<<(Cc * Cc + 255) / 256, 256, 0, stream>>>(Wq, wqt, Cc, Cc);
  transpose_cvt<<<(Cc * 512 + 255) / 256, 256, 0, stream>>>(Wk, wkt, Cc, 512);
  transpose_cvt<<<(Cc * 512 + 255) / 256, 256, 0, stream>>>(Wv, wvt, Cc, 512);
  transpose_cvt<<<(Cc * Cc + 255) / 256, 256, 0, stream>>>(Wpr, wpt, Cc, Cc);

  // QKV projections (WMMA): each wave covers 32 rows -> grid.x = M/256
  gemm16<<<dim3(Mc / 256, 1024 / 64), 256, 0, stream>>>(xh, wqt, qf, Mc, 1024, Cc);
  gemm16<<<dim3(Mc / 256, 512 / 64),  256, 0, stream>>>(xh, wkt, kf, Mc, 512, Cc);
  gemm16<<<dim3(Mc / 256, 512 / 64),  256, 0, stream>>>(xh, wvt, vf, Mc, 512, Cc);

  // gate + value-embed + rope + rmsnorm
  pointwise<<<Mc, 128, 0, stream>>>(x, ve, cosb, sinb, Wg, qf, kf, vf, qhp, khp, vtp);

  // flash attention (WMMA)
  attn<<<dim3(Tc / 16 / 8, Bc * NHc), 256, 0, stream>>>(qhp, khp, vtp, yh);

  // output projection (WMMA) -> fp32 result
  gemm16<<<dim3(Mc / 256, Cc / 64), 256, 0, stream>>>(yh, wpt, (float*)d_out, Mc, Cc, Cc);
}
